// MobiusSplineFlow_67817533604252
// MI455X (gfx1250) — matrix operand order repaired and checked by reading, not compiled, not run on hardware
//
#include <hip/hip_runtime.h>
#include <hip/hip_bf16.h>
#include <math.h>

// ---------------------------------------------------------------------------
// MobiusSplineFlow for MI455X (gfx1250, wave32).
//  - Layer GEMMs via v_wmma_f32_16x16x32_f16 (transposed orientation so the
//    activation is the B operand: lane = point, halves = K).
//  - W2^T / W3^T pre-swizzled into LDS in exact A-fragment layout once/block.
//  - Biases folded into the WMMA C operand.
//  - h2 transpose via wave-private LDS + s_wait_dscnt (CDNA5 split counters).
//  - Mobius transcendentals on VALU (co-executes with XDL WMMA on CDNA5);
//    2 lanes per point, combined with ds_swizzle SWAPX16.
// ---------------------------------------------------------------------------

typedef __attribute__((ext_vector_type(16))) _Float16 v16h;
typedef __attribute__((ext_vector_type(8)))  _Float16 v8h;
typedef __attribute__((ext_vector_type(8)))  float    v8f;

union F16x16 { v16h v; _Float16 h[16]; };
union F16x8  { v8h  v; _Float16 h[8];  };
union F32x8  { v8f  v; float    f[8];  };

#define NPTS 1000000
#define KS   64
#define TWO_PI_F 6.283185307179586f

__device__ __forceinline__ float lane_swap16(float x) {
  // group-of-32 swizzle, xor_mask=0x10, and_mask=0x1f  -> lane ^ 16
  return __int_as_float(__builtin_amdgcn_ds_swizzle(__float_as_int(x), 0x401F));
}

__global__ __launch_bounds__(256)
void mobius_spline_kernel(const float* __restrict__ r,  const float* __restrict__ z,
                          const float* __restrict__ tw, const float* __restrict__ thh,
                          const float* __restrict__ td,
                          const float* __restrict__ W1, const float* __restrict__ b1,
                          const float* __restrict__ W2, const float* __restrict__ b2,
                          const float* __restrict__ W3, const float* __restrict__ b3,
                          float* __restrict__ out)
{
  // ---- LDS -----------------------------------------------------------------
  __shared__ _Float16 w2frag[8][32][16] __attribute__((aligned(32))); // 8 KB
  __shared__ _Float16 w3frag[6][32][16] __attribute__((aligned(32))); // 6 KB
  __shared__ _Float16 trans[8][16][64]  __attribute__((aligned(32))); // 16 KB (per-wave h2^T)
  __shared__ float    thetaBuf[8][16][48] __attribute__((aligned(32))); // 24 KB
  __shared__ float    xks[KS + 1], yks[KS + 1], dervs[KS + 1];

  const int tid  = threadIdx.x;
  const int lane = tid & 31;
  const int wv   = tid >> 5;
  const int m16  = lane & 15;   // point-in-tile (B/C/D lane index)
  const int kh   = lane >> 4;   // half-wave id

  // ---- stage W2^T and W3^T as A-fragments (f32 -> f16), once per block ----
  // A-fragment k order used here (contiguous, same as B): halves 0..15 -> K = 16*kh + i
  for (int idx = tid; idx < 8 * 32 * 16; idx += 256) {
    int f = idx >> 9, rem = idx & 511, l = rem >> 4, i = rem & 15;
    int mt = f >> 1, kc = f & 1, lm = l & 15, lk = l >> 4;
    int k = 32 * kc + 16 * lk + i;
    w2frag[f][l][i] = (_Float16)W2[k * 64 + (16 * mt + lm)];  // W2^T[m,k] = W2[k,m]
  }
  for (int idx = tid; idx < 6 * 32 * 16; idx += 256) {
    int f = idx >> 9, rem = idx & 511, l = rem >> 4, i = rem & 15;
    int mt = f >> 1, kc = f & 1, lm = l & 15, lk = l >> 4;
    int k = 32 * kc + 16 * lk + i;
    w3frag[f][l][i] = (_Float16)W3[k * 48 + (16 * mt + lm)];  // W3^T[m,k] = W3[k,m]
  }
  // ---- spline knot tables (softmax+cumsum+softplus), once per block -------
  if (tid == 0) {
    float sw = 0.f, sh = 0.f;
    for (int i = 0; i < KS; ++i) { sw += __expf(tw[i]); sh += __expf(thh[i]); }
    float cx = 0.f, cy = 0.f;
    xks[0] = -1.f; yks[0] = -1.f;
    for (int i = 0; i < KS; ++i) {
      cx += __expf(tw[i])  / sw * 2.f;
      cy += __expf(thh[i]) / sh * 2.f;
      xks[i + 1] = cx - 1.f; yks[i + 1] = cy - 1.f;
    }
    dervs[0] = 1.f; dervs[KS] = 1.f;
    for (int i = 0; i < KS - 1; ++i) dervs[i + 1] = log1pf(__expf(td[i])) + 0.001f;
  }
  __syncthreads();

  // ---- per-lane loop invariants (registers) -------------------------------
  F16x16 W1h[2], B1h[2];        // layer-1 weights/bias in B-fragment half order
#pragma unroll
  for (int kc = 0; kc < 2; ++kc)
#pragma unroll
    for (int i = 0; i < 16; ++i) {
      int k = 32 * kc + 16 * kh + i;
      W1h[kc].h[i] = (_Float16)W1[k];
      B1h[kc].h[i] = (_Float16)b1[k];
    }
  F32x8 b2r[4];                 // bias as WMMA C operand (C layout)
#pragma unroll
  for (int mt = 0; mt < 4; ++mt)
#pragma unroll
    for (int j = 0; j < 8; ++j) b2r[mt].f[j] = b2[16 * mt + 8 * kh + j];
  F32x8 b3r[3];
#pragma unroll
  for (int mt = 0; mt < 3; ++mt)
#pragma unroll
    for (int j = 0; j < 8; ++j) b3r[mt].f[j] = b3[16 * mt + 8 * kh + j];

  // ---- grid-stride over 16-point tiles, one tile per wave -----------------
  const int ntiles  = NPTS / 16;
  const int gstride = gridDim.x * 8;
  for (int t = blockIdx.x * 8 + wv; t < ntiles; t += gstride) {
    const int base = t * 16;
    const float rv = r[base + m16];

    // layer 1 directly in B-fragment layout: h1[n,k] = relu(r[n]*W1[k]+b1[k])
    F16x16 bf0, bf1;
#pragma unroll
    for (int i = 0; i < 16; ++i) {
      float h0 = fmaxf(rv * (float)W1h[0].h[i] + (float)B1h[0].h[i], 0.f);
      float h1 = fmaxf(rv * (float)W1h[1].h[i] + (float)B1h[1].h[i], 0.f);
      bf0.h[i] = (_Float16)h0;
      bf1.h[i] = (_Float16)h1;
    }

    // layer 2: h2^T = relu(W2^T @ h1^T + b2); D lane = point, VGPR = feature
#pragma unroll
    for (int mt = 0; mt < 4; ++mt) {
      v16h a0 = *(const v16h*)(&w2frag[2 * mt + 0][lane][0]);
      v16h a1 = *(const v16h*)(&w2frag[2 * mt + 1][lane][0]);
      F32x8 acc; acc.v = b2r[mt].v;
      acc.v = __builtin_amdgcn_wmma_f32_16x16x32_f16(false, a0, false, bf0.v,
                                                     (short)0, acc.v, false, false);
      acc.v = __builtin_amdgcn_wmma_f32_16x16x32_f16(false, a1, false, bf1.v,
                                                     (short)0, acc.v, false, false);
      F16x8 hh;
#pragma unroll
      for (int j = 0; j < 8; ++j) hh.h[j] = (_Float16)fmaxf(acc.f[j], 0.f);
      // wave-private transpose buffer: [point][feature], 16B-aligned b128 store
      *(v8h*)(&trans[wv][m16][16 * mt + 8 * kh]) = hh.v;
    }
    asm volatile("s_wait_dscnt 0x0" ::: "memory");   // DS in-order within wave

    // reload h2 as B fragments for layer 3 (lane = point, halves = K)
    v16h h2b0 = *(const v16h*)(&trans[wv][m16][16 * kh]);
    v16h h2b1 = *(const v16h*)(&trans[wv][m16][32 + 16 * kh]);

    // layer 3: theta^T = W3^T @ h2^T + b3
#pragma unroll
    for (int mt = 0; mt < 3; ++mt) {
      v16h a0 = *(const v16h*)(&w3frag[2 * mt + 0][lane][0]);
      v16h a1 = *(const v16h*)(&w3frag[2 * mt + 1][lane][0]);
      F32x8 acc; acc.v = b3r[mt].v;
      acc.v = __builtin_amdgcn_wmma_f32_16x16x32_f16(false, a0, false, h2b0,
                                                     (short)0, acc.v, false, false);
      acc.v = __builtin_amdgcn_wmma_f32_16x16x32_f16(false, a1, false, h2b1,
                                                     (short)0, acc.v, false, false);
      *(v8f*)(&thetaBuf[wv][m16][16 * mt + 8 * kh]) = acc.v;   // 32B-aligned
    }
    asm volatile("s_wait_dscnt 0x0" ::: "memory");

    // ---- Mobius elementwise: lanes (p, p+16) each do 8 of 16 centers ------
    const float zv = z[base + m16];
    const float cz = __cosf(zv), sz = __sinf(zv);
    float sum_e = 0.f, sum_et = 0.f, sum_ec = 0.f;
    const float* th = &thetaBuf[wv][m16][0];
#pragma unroll
    for (int j = 0; j < 8; ++j) {
      int km = 8 * kh + j;
      float lw = th[km];
      float wa = th[16 + 2 * km], wb = th[17 + 2 * km];
      float nrm = __fsqrt_rn(wa * wa + wb * wb);
      float s   = 0.99f / (1.f + nrm);
      float wx = s * wa, wy = s * wb;
      float one_m = 1.f - (wx * wx + wy * wy);
      // h(z)
      float dx = cz - wx, dy = sz - wy;
      float dn2 = dx * dx + dy * dy;
      float cf  = one_m / dn2;
      float rad = atan2f(cf * dy - wy, cf * dx - wx);
      // h(1,0)
      float d0x = 1.f - wx, d0y = -wy;
      float c0  = one_m / (d0x * d0x + d0y * d0y);
      float shf = atan2f(c0 * d0y - wy, c0 * d0x - wx);
      float tx = rad - shf;
      tx = (tx >= 0.f) ? tx : tx + TWO_PI_F;
      float e = __expf(lw);
      sum_e  += e;
      sum_et += e * tx;
      sum_ec += e * cf;       // |dh| = (1-|w|^2)/|diff|^2 (dz is unit, reflection)
    }
    sum_e  += lane_swap16(sum_e);
    sum_et += lane_swap16(sum_et);
    sum_ec += lane_swap16(sum_ec);
    const float tzv = sum_et / sum_e;
    const float dtz = sum_ec / sum_e;

    // ---- rational-quadratic spline on r -----------------------------------
    const float x = rv;
    int lo = 0, hi = KS + 1;                     // searchsorted(left)
    while (lo < hi) { int mid = (lo + hi) >> 1; if (xks[mid] < x) lo = mid + 1; else hi = mid; }
    int k = lo;
    if (k == 0)      k = 1;
    if (k == KS + 1) k = KS;
    k -= 1;
    const float x_k = xks[k],   x_nk = xks[k + 1];
    const float y_k = yks[k],   y_nk = yks[k + 1];
    const float d_k = dervs[k], d_nk = dervs[k + 1];
    const float s_k = (y_nk - y_k) / (x_nk - x_k);
    const float eps = (x - x_k) / (x_nk - x_k);
    const float ome = 1.f - eps;
    const float den = s_k + (d_nk + d_k - 2.f * s_k) * eps * ome;
    const float trv = y_k + (y_nk - y_k) * (s_k * eps * eps + d_k * eps * ome) / den;
    const float dtr = s_k * s_k * (d_nk * eps * eps + 2.f * s_k * eps * ome + d_k * ome * ome)
                      / (den * den);

    if (kh == 0) {
      out[base + m16]            = trv;                       // tr
      out[NPTS + base + m16]     = tzv;                       // tz
      out[2 * NPTS + base + m16] = __logf(dtr) + __logf(dtz); // ldj
    }
  }
}

extern "C" void kernel_launch(void* const* d_in, const int* in_sizes, int n_in,
                              void* d_out, int out_size, void* d_ws, size_t ws_size,
                              hipStream_t stream) {
  (void)in_sizes; (void)n_in; (void)out_size; (void)d_ws; (void)ws_size;
  mobius_spline_kernel<<<1024, 256, 0, stream>>>(
      (const float*)d_in[0],  (const float*)d_in[1],  (const float*)d_in[2],
      (const float*)d_in[3],  (const float*)d_in[4],  (const float*)d_in[5],
      (const float*)d_in[6],  (const float*)d_in[7],  (const float*)d_in[8],
      (const float*)d_in[9],  (const float*)d_in[10], (float*)d_out);
}
